// MC_GRU_87333864996865
// MI455X (gfx1250) — matrix operand order repaired and checked by reading, not compiled, not run on hardware
//
#include <hip/hip_runtime.h>
#include <hip/hip_bf16.h>

// ---------------------------------------------------------------------------
// MC-GRU scan for MI455X (gfx1250, wave32, WMMA).
//
//   - grid = B/16 = 16 persistent workgroups; WG bt owns batch rows
//     [16*bt, 16*bt+16) for the whole T=1000 scan (no inter-WG sync).
//   - 256 threads = 8 waves; wave w owns output columns [64w, 64w+64)
//     (4 WMMA N-tiles) of z / r / h_tilde / h.
//   - All three GEMMs are K=512 (h part); x, St*k, Mass rows are rank-1
//     corrections applied in VALU (IN==1).
//   - h lives in:  (a) f32 registers in WMMA C-layout (per-wave 64 cols),
//                  (b) bf16 row-major [16][512] LDS image => A-matrix source.
//   - Weights are packed ONCE into bf16 B-matrix tile layout (32B/lane/tile).
//   - B-tile stream is software-pipelined: double-buffered over K-tiles so
//     one partial s_wait_loadcnt covers 8 back-to-back v_wmma ops and the
//     next iteration's 16 loads are already in flight.
// ---------------------------------------------------------------------------

typedef __attribute__((ext_vector_type(16))) __bf16 v16bf;
typedef __attribute__((ext_vector_type(8)))  __bf16 v8bf;
typedef __attribute__((ext_vector_type(8)))  float  v8f;

#define B_ALL   256
#define T_STEPS 1000
#define HDIM    512
#define KT_N    16          // 512 / 32  K-tiles
#define LDA     520         // bf16 row stride for LDS h images (pad, 16B-mult)
#define PER_MTX (512 * 512) // packed bf16 elems per weight matrix

__device__ __forceinline__ float sigmoid_f(float x) {
  return __builtin_amdgcn_rcpf(1.0f + __builtin_amdgcn_exp2f(-1.4426950408889634f * x));
}
__device__ __forceinline__ float tanh_f(float x) {
  float s = __builtin_amdgcn_rcpf(1.0f + __builtin_amdgcn_exp2f(-2.8853900817779268f * x));
  return 2.0f * s - 1.0f;
}

__device__ __forceinline__ v16bf load_a_tile(const __bf16* ap) {
  v8bf alo = *(const v8bf*)ap;          // K run 0..7  (or 8..15)
  v8bf ahi = *(const v8bf*)(ap + 16);   // K run 16..23 (or 24..31)
  return __builtin_shufflevector(alo, ahi,
           0,1,2,3,4,5,6,7,8,9,10,11,12,13,14,15);
}

// ---------------------------------------------------------------------------
// Prologue: convert W[k+1][n] (f32) -> bf16 in B-matrix tile layout.
// Tile (ntile,ktile) is 32x16 (KxN), 1 KB: lane = (n%16)+16*h,
// lane's 16 bf16 are K = ktile*32 + 8h + 0..7, then ktile*32 + 16 + 8h + 0..7.
// ---------------------------------------------------------------------------
__global__ __launch_bounds__(256) void pack_weights(
    const float* __restrict__ Wz, const float* __restrict__ Wr,
    const float* __restrict__ Wh, __bf16* __restrict__ ws) {
  int gid  = blockIdx.x * blockDim.x + threadIdx.x;   // [0, 3*512*512)
  int mtx  = gid / PER_MTX;
  int e    = gid - mtx * PER_MTX;
  int tidx = e >> 9;          // tile index = ntile*16 + ktile
  int r    = e & 511;
  int lane = r >> 4;
  int i    = r & 15;
  int ntile = tidx >> 4;
  int ktile = tidx & 15;
  int n = ntile * 16 + (lane & 15);
  int k = ktile * 32 + ((lane >> 4) << 3) + (i & 7) + ((i >> 3) << 4);
  const float* W = (mtx == 0) ? Wz : (mtx == 1) ? Wr : Wh;
  ws[gid] = (__bf16)W[(size_t)(k + 1) * HDIM + n];    // row 0 is the x-row
}

// ---------------------------------------------------------------------------
// Main persistent scan kernel.
// ---------------------------------------------------------------------------
__global__ __launch_bounds__(256) void mc_gru_scan(
    const float* __restrict__ x_in,   // [B, T]  (IN == 1)
    const float* __restrict__ St,     // [B]
    const float* __restrict__ Mass,   // [B]
    const float* __restrict__ Wz, const float* __restrict__ bz,
    const float* __restrict__ Wr, const float* __restrict__ br,
    const float* __restrict__ Wh, const float* __restrict__ bh,
    const float* __restrict__ Wk, const float* __restrict__ bk,
    const __bf16* __restrict__ Wpk,   // packed bf16: [3][512*512]
    float* __restrict__ out)          // [B, T, H]
{
  __shared__ __bf16 hA [16 * LDA];    // h_{t-1}, row-major bf16 (A source)
  __shared__ __bf16 hrA[16 * LDA];    // h*r,     row-major bf16 (A source)
  __shared__ float  xv[16], stv[16], massv[16], kv[16], kacc[16];

  const int tid  = threadIdx.x;
  const int wave = tid >> 5;
  const int lane = tid & 31;
  const int hlf  = lane >> 4;         // 0: C rows 0-7 / A K-run lo, 1: rows 8-15
  const int ln   = lane & 15;
  const int bt   = blockIdx.x;        // batch tile

  // ---- init LDS state -----------------------------------------------------
  for (int i = tid; i < 16 * LDA; i += 256) { hA[i] = (__bf16)0.0f; hrA[i] = (__bf16)0.0f; }
  if (tid < 16) {
    int b = bt * 16 + tid;
    kv[tid]    = 0.0f;
    kacc[tid]  = 0.0f;
    stv[tid]   = St[b];
    massv[tid] = Mass[b];
    xv[tid]    = x_in[(size_t)b * T_STEPS];   // x at t=0
  }

  // ---- per-lane constants (4 N-tiles per wave) ----------------------------
  float wz0[4], wr0[4], wh0[4], whk[4], whm[4], bzv[4], brv[4], bhv[4], wkvv[4];
#pragma unroll
  for (int j = 0; j < 4; ++j) {
    int n = (wave * 4 + j) * 16 + ln;
    wz0[j] = Wz[n];                       // x-row of Wz
    wr0[j] = Wr[n];
    wh0[j] = Wh[n];                       // x-row of Wh
    whk[j] = Wh[(size_t)513 * HDIM + n];  // St*k row
    whm[j] = Wh[(size_t)514 * HDIM + n];  // Mass row
    bzv[j] = bz[n];  brv[j] = br[n];  bhv[j] = bh[n];
    wkvv[j] = Wk[n];                      // Wk is [513,1]
  }
  const float wk_k = Wk[512];
  const float bk_s = bk[0];

  float h_reg[4][8];                      // f32 h, C-layout, this wave's cols
#pragma unroll
  for (int j = 0; j < 4; ++j)
#pragma unroll
    for (int v = 0; v < 8; ++v) h_reg[j][v] = 0.0f;

  // per-lane base pointers: tile (j,kt) at + j*16KB + kt*1KB (imm-foldable)
  const __bf16* WzL = Wpk + (size_t)(wave * 4) * 16 * 512 + (lane << 4);
  const __bf16* WrL = WzL + (size_t)PER_MTX;
  const __bf16* WhL = WzL + (size_t)2 * PER_MTX;
  const int aoff = ln * LDA + (hlf << 3);  // per-lane A base (bf16 elems)

  __syncthreads();

  // ======================= sequential scan over time =======================
  for (int t = 0; t < T_STEPS; ++t) {
    // per-step per-lane scalars for the rank-1 input rows
    float xr[8], skr[8], mar[8];
#pragma unroll
    for (int v = 0; v < 8; ++v) {
      int m  = v + (hlf << 3);
      xr[v]  = xv[m];
      skr[v] = stv[m] * kv[m];
      mar[v] = massv[m];
    }

    // ---- phase A: z_pre / r_pre GEMMs (K=512, pipelined B stream) ---------
    v8f accz[4], accr[4];
#pragma unroll
    for (int j = 0; j < 4; ++j)
#pragma unroll
      for (int v = 0; v < 8; ++v) {
        accz[j][v] = xr[v] * wz0[j] + bzv[j];
        accr[j][v] = xr[v] * wr0[j] + brv[j];
      }

    {
      v16bf bzb[2][4], brb[2][4], ab[2];
      // prologue: K-tile 0
      ab[0] = load_a_tile(&hA[aoff]);
#pragma unroll
      for (int j = 0; j < 4; ++j) {
        bzb[0][j] = *(const v16bf*)(WzL + j * 8192);
        brb[0][j] = *(const v16bf*)(WrL + j * 8192);
      }
#pragma unroll 2
      for (int kt = 0; kt < KT_N; ++kt) {
        const int p  = kt & 1;
        const int np = p ^ 1;
        if (kt + 1 < KT_N) {            // issue next K-tile's loads first
          ab[np] = load_a_tile(&hA[aoff + (kt + 1) * 32]);
#pragma unroll
          for (int j = 0; j < 4; ++j) {
            bzb[np][j] = *(const v16bf*)(WzL + j * 8192 + (kt + 1) * 512);
            brb[np][j] = *(const v16bf*)(WrL + j * 8192 + (kt + 1) * 512);
          }
        }
#pragma unroll
        for (int j = 0; j < 4; ++j) {
          accz[j] = __builtin_amdgcn_wmma_f32_16x16x32_bf16(
              false, ab[p], false, bzb[p][j], (short)0, accz[j], false, false);
          accr[j] = __builtin_amdgcn_wmma_f32_16x16x32_bf16(
              false, ab[p], false, brb[p][j], (short)0, accr[j], false, false);
        }
      }
    }

    // ---- phase B: gates, write h*r (bf16) to LDS --------------------------
#pragma unroll
    for (int j = 0; j < 4; ++j) {
      int n = (wave * 4 + j) * 16 + ln;
#pragma unroll
      for (int v = 0; v < 8; ++v) {
        int m = v + (hlf << 3);
        float z  = sigmoid_f(accz[j][v]);
        float r  = sigmoid_f(accr[j][v]);
        accz[j][v] = z;                                   // keep z
        hrA[m * LDA + n] = (__bf16)(h_reg[j][v] * r);
      }
    }
    __syncthreads();   // (1) hrA complete; all hA reads of this step done

    // ---- phase C: h_tilde GEMM (K=512 over h*r, pipelined) ----------------
    v8f acch[4];
#pragma unroll
    for (int j = 0; j < 4; ++j)
#pragma unroll
      for (int v = 0; v < 8; ++v)
        acch[j][v] = xr[v] * wh0[j] + skr[v] * whk[j] + mar[v] * whm[j] + bhv[j];

    {
      v16bf bhb[2][4], ab[2];
      ab[0] = load_a_tile(&hrA[aoff]);
#pragma unroll
      for (int j = 0; j < 4; ++j)
        bhb[0][j] = *(const v16bf*)(WhL + j * 8192);
#pragma unroll 2
      for (int kt = 0; kt < KT_N; ++kt) {
        const int p  = kt & 1;
        const int np = p ^ 1;
        if (kt + 1 < KT_N) {
          ab[np] = load_a_tile(&hrA[aoff + (kt + 1) * 32]);
#pragma unroll
          for (int j = 0; j < 4; ++j)
            bhb[np][j] = *(const v16bf*)(WhL + j * 8192 + (kt + 1) * 512);
        }
#pragma unroll
        for (int j = 0; j < 4; ++j)
          acch[j] = __builtin_amdgcn_wmma_f32_16x16x32_bf16(
              false, ab[p], false, bhb[p][j], (short)0, acch[j], false, false);
      }
    }

    // ---- phase D: h_new, store out + hA, k partials -----------------------
    float pk[8];
#pragma unroll
    for (int v = 0; v < 8; ++v) pk[v] = 0.0f;

#pragma unroll
    for (int j = 0; j < 4; ++j) {
      int n = (wave * 4 + j) * 16 + ln;
#pragma unroll
      for (int v = 0; v < 8; ++v) {
        int m = v + (hlf << 3);
        float z  = accz[j][v];
        float ht = tanh_f(acch[j][v]);
        float hn = (1.0f - z) * h_reg[j][v] + z * ht;
        h_reg[j][v] = hn;
        out[((size_t)(bt * 16 + m) * T_STEPS + t) * HDIM + n] = hn;
        hA[m * LDA + n] = (__bf16)hn;                    // h_t for next step
        pk[v] += hn * wkvv[j];
      }
    }

    // reduce k partials across the 16 lanes of each half (same row set)
#pragma unroll
    for (int mask = 1; mask <= 8; mask <<= 1)
#pragma unroll
      for (int v = 0; v < 8; ++v) pk[v] += __shfl_xor(pk[v], mask, 32);
    if (ln == 0) {
#pragma unroll
      for (int v = 0; v < 8; ++v) atomicAdd(&kacc[v + (hlf << 3)], pk[v]);
    }

    __syncthreads();   // (2) kacc + hA complete

    if (tid < 16) {
      float kn = sigmoid_f(kacc[tid] + kv[tid] * wk_k + bk_s);
      kv[tid]   = kn;
      kacc[tid] = 0.0f;
      if (t + 1 < T_STEPS)
        xv[tid] = x_in[(size_t)(bt * 16 + tid) * T_STEPS + (t + 1)];
    }
    __syncthreads();   // (3) kv / xv / kacc ready for next step
  }
}

extern "C" void kernel_launch(void* const* d_in, const int* in_sizes, int n_in,
                              void* d_out, int out_size, void* d_ws, size_t ws_size,
                              hipStream_t stream) {
  const float* x_in = (const float*)d_in[0];
  const float* St   = (const float*)d_in[1];
  const float* Mass = (const float*)d_in[2];
  const float* Wz   = (const float*)d_in[3];
  const float* bz   = (const float*)d_in[4];
  const float* Wr   = (const float*)d_in[5];
  const float* br   = (const float*)d_in[6];
  const float* Wh   = (const float*)d_in[7];
  const float* bh   = (const float*)d_in[8];
  const float* Wk   = (const float*)d_in[9];
  const float* bk   = (const float*)d_in[10];

  __bf16* wpk = (__bf16*)d_ws;   // needs 3*512*512*2 = 1.5 MB of scratch

  pack_weights<<<(3 * PER_MTX) / 256, 256, 0, stream>>>(Wz, Wr, Wh, wpk);
  mc_gru_scan<<<B_ALL / 16, 256, 0, stream>>>(
      x_in, St, Mass, Wz, bz, Wr, br, Wh, bh, Wk, bk, wpk, (float*)d_out);
}
